// Embedding_87084756893716
// MI455X (gfx1250) — matrix operand order repaired and checked by reading, not compiled, not run on hardware
//
#include <hip/hip_runtime.h>

typedef __attribute__((ext_vector_type(16))) __bf16        v16bf;
typedef __attribute__((ext_vector_type(8)))  float         v8f;
typedef __attribute__((ext_vector_type(8)))  unsigned int  v8u;

#define UNITS   384
#define NT      24          // 384 / 16 N-tiles
#define HSTRIDE 392         // 384 + 8 bf16 pad  -> conflict-free ds_read_b128
#define XSTRIDE 104         // 96 + 8 bf16 pad
#define SEQ     64
#define NB      1024

// ---- packed-weight layout offsets (dwords), relative to wpack base ----
// w1 tiles: lips 3kt, lh 2kt, rh 2kt, pose 1kt  (x 24 nt, 256 dwords/tile)
#define OFF_LIPS1 0
#define OFF_LH1   18432      // 72*256
#define OFF_RH1   30720      // 120*256
#define OFF_POSE1 43008      // 168*256
#define OFF_LIPS2 49152      // 192*256
#define OFF_LH2   122880
#define OFF_RH2   196608
#define OFF_POSE2 270336
#define OFF_FC1   344064
#define OFF_FC2   417792
#define WPACK_DWORDS 491520

__device__ __forceinline__ unsigned short f2bf(float f) {
    unsigned int x = __float_as_uint(f);
    return (unsigned short)((x + 0x7FFFu + ((x >> 16) & 1u)) >> 16);  // RNE
}
__device__ __forceinline__ unsigned int pack2bf(float f0, float f1) {
    return (unsigned int)f2bf(f0) | ((unsigned int)f2bf(f1) << 16);
}

// A fragment (16x32 bf16) per ISA layout: lane m=lane&15, hi=lane>>4.
// VGPR0..3 hold K = hi*8 + 0..7, VGPR4..7 hold K = 16 + hi*8 + 0..7
// -> exactly two 16B LDS reads from row-major [16][stride] bf16.
__device__ __forceinline__ v16bf load_a_lds(const unsigned short* base) {
    uint4 lo = *(const uint4*)(base);        // K offsets hi*8 .. +7
    uint4 hi = *(const uint4*)(base + 16);   // K offsets 16+hi*8 .. +7
    v8u u;
    u[0] = lo.x; u[1] = lo.y; u[2] = lo.z; u[3] = lo.w;
    u[4] = hi.x; u[5] = hi.y; u[6] = hi.z; u[7] = hi.w;
    return __builtin_bit_cast(v16bf, u);
}

// B fragment pre-packed in global memory: tile = 256 dwords, lane-major
// (lane*8 + j), so each lane does two coalesced b128 loads.
__device__ __forceinline__ v16bf load_b_glb(const unsigned int* base) {
    uint4 lo = *(const uint4*)(base);
    uint4 hi = *(const uint4*)(base + 4);
    v8u u;
    u[0] = lo.x; u[1] = lo.y; u[2] = lo.z; u[3] = lo.w;
    u[4] = hi.x; u[5] = hi.y; u[6] = hi.z; u[7] = hi.w;
    return __builtin_bit_cast(v16bf, u);
}

// One K-step of the wave's 16x48 output: 1 A fragment (LDS, immediate
// offset) x 3 contiguous B fragments (global, immediate offsets 0/1/2 KB).
__device__ __forceinline__ void gemm_step(const unsigned short* __restrict__ abase,
                                          int aoff,
                                          const unsigned int* __restrict__ bbase,
                                          v8f acc[3]) {
    v16bf a = load_a_lds(abase + aoff);
#pragma unroll
    for (int t = 0; t < 3; ++t) {
        v16bf bm = load_b_glb(bbase + t * 256);
        acc[t] = __builtin_amdgcn_wmma_f32_16x16x32_bf16(
            false, a, false, bm, (short)0, acc[t], false, false);
    }
}

// ---------------- prep kernels ----------------

// Pack one [Ksrc, 384] f32 weight into bf16 B-fragment tiles.
// grid = kt_tiles*24 blocks, 32 threads (one wave) per tile.
__global__ void pack_weights(const float* __restrict__ src, int Ksrc,
                             unsigned int* __restrict__ dst) {
    const int tile  = blockIdx.x;           // kt*24 + nt
    const int kt    = tile / NT;
    const int nt    = tile % NT;
    const int lane  = threadIdx.x;
    const int khalf = lane >> 4;
    const int n     = nt * 16 + (lane & 15);
    unsigned int* out = dst + (size_t)tile * 256 + lane * 8;
#pragma unroll
    for (int j = 0; j < 8; ++j) {
        int k0 = kt * 32 + khalf * 8 + (j >> 2) * 16 + 2 * (j & 3);
        float f0 = (k0     < Ksrc) ? src[(size_t)k0       * UNITS + n] : 0.f;
        float f1 = (k0 + 1 < Ksrc) ? src[(size_t)(k0 + 1) * UNITS + n] : 0.f;
        out[j] = pack2bf(f0, f1);
    }
}

__global__ void prep_softmax(const float* __restrict__ lw, float* __restrict__ wsm) {
    float a0 = lw[0], a1 = lw[1], a2 = lw[2], a3 = lw[3];
    float m  = fmaxf(fmaxf(a0, a1), fmaxf(a2, a3));
    float e0 = __expf(a0 - m), e1 = __expf(a1 - m), e2 = __expf(a2 - m), e3 = __expf(a3 - m);
    float inv = 1.0f / (e0 + e1 + e2 + e3);
    wsm[0] = e0 * inv; wsm[1] = e1 * inv; wsm[2] = e2 * inv; wsm[3] = e3 * inv;
}

// Per-row pos_emb index (one block of 64 threads per batch element).
__global__ void prep_idx(const float* __restrict__ nefi, int* __restrict__ idx) {
    __shared__ float red[SEQ];
    const int b = blockIdx.x, s = threadIdx.x;
    float v = nefi[b * SEQ + s];
    red[s] = v;
    __syncthreads();
    for (int off = 32; off > 0; off >>= 1) {
        if (s < off) red[s] = fmaxf(red[s], red[s + off]);
        __syncthreads();
    }
    float rmax = red[0];
    int id = (int)floorf(v / rmax) * SEQ;
    if (v == -1.0f) id = SEQ;
    idx[b * SEQ + s] = id;
}

// ---------------- fused main kernel ----------------
__launch_bounds__(256)
__global__ void fused_embed(const float* __restrict__ lips,
                            const float* __restrict__ lh,
                            const float* __restrict__ rh,
                            const float* __restrict__ pose,
                            const unsigned int* __restrict__ wpack,
                            const float* __restrict__ wsm,
                            const int* __restrict__ posidx,
                            const float* __restrict__ pos_emb,
                            float* __restrict__ out) {
    __shared__ unsigned short xbuf[16 * XSTRIDE];   // branch inputs (bf16, zero padded)
    __shared__ unsigned short hbuf[16 * HSTRIDE];   // 16 x 384 bf16 activations
    __shared__ float psum[256];
    __shared__ float flag[16];
    __shared__ int   pid_s[16];

    const int tid  = threadIdx.x;
    const int lane = tid & 31;
    const int wave = tid >> 5;
    const int m16  = lane & 15;
    const int hi   = lane >> 4;

    const int blk = blockIdx.x;
    const int b   = blk >> 2;
    const int s0  = (blk & 3) << 4;
    const long rowBase = (long)b * SEQ + s0;     // first of 16 output rows

    // loop-invariant fragment base pointers (immediate offsets do the rest)
    const unsigned short* xab = xbuf + m16 * XSTRIDE + hi * 8;
    const unsigned short* hab = hbuf + m16 * HSTRIDE + hi * 8;
    const int bwoff = wave * 768 + lane * 8;     // wave's 3 contiguous B tiles

    const v8f zero = {0.f, 0.f, 0.f, 0.f, 0.f, 0.f, 0.f, 0.f};

    const int   Farr[4]  = {80, 42, 42, 20};
    const int   KT1[4]   = {3, 2, 2, 1};
    const int   w1off[4] = {OFF_LIPS1, OFF_LH1, OFF_RH1, OFF_POSE1};
    const int   w2off[4] = {OFF_LIPS2, OFF_LH2, OFF_RH2, OFF_POSE2};
    const float* xin[4]  = {lips + rowBase * 80, lh + rowBase * 42,
                            rh  + rowBase * 42, pose + rowBase * 20};

    v8f xacc[3];
    xacc[0] = zero; xacc[1] = zero; xacc[2] = zero;

    for (int br = 0; br < 4; ++br) {
        __syncthreads();
        // ---- stage branch input -> xbuf (bf16, K zero-padded to 96) + flags
        {
            const int   F   = Farr[br];
            const float* src = xin[br];
            const int row = tid >> 4, cs = tid & 15;
            float ps = 0.f;
            for (int c = cs; c < 96; c += 16) {
                float v = 0.f;
                if (c < F) { v = src[row * F + c]; ps += v; }
                xbuf[row * XSTRIDE + c] = f2bf(v);
            }
            psum[tid] = ps;
        }
        __syncthreads();
        if (tid < 16) {
            float s = 0.f;
            for (int i = 0; i < 16; ++i) s += psum[tid * 16 + i];
            flag[tid] = (s == 0.f) ? 0.f : 1.f;
        }
        __syncthreads();

        // ---- GEMM1: h = relu(x @ w1), per-wave 3 N-tiles
        v8f acc[3];
        acc[0] = zero; acc[1] = zero; acc[2] = zero;
        {
            const unsigned int* bb = wpack + w1off[br] + bwoff;
            for (int kt = 0; kt < KT1[br]; ++kt, bb += NT * 256)
                gemm_step(xab, kt * 32, bb, acc);
        }
        // relu + empty-row mask -> hbuf (bf16)
#pragma unroll
        for (int t = 0; t < 3; ++t) {
            const int nt = wave * 3 + t;
#pragma unroll
            for (int v = 0; v < 8; ++v) {
                const int m = v + 8 * hi;
                float h = fmaxf(acc[t][v], 0.f) * flag[m];
                hbuf[m * HSTRIDE + nt * 16 + m16] = f2bf(h);
            }
        }
        __syncthreads();

        // ---- GEMM2: e = h @ w2 ; xacc += softmax_w[br] * e
        v8f acc2[3];
        acc2[0] = zero; acc2[1] = zero; acc2[2] = zero;
        {
            const unsigned int* bb = wpack + w2off[br] + bwoff;
#pragma unroll 4
            for (int kt = 0; kt < 12; ++kt, bb += NT * 256)
                gemm_step(hab, kt * 32, bb, acc2);
        }
        const float wk = wsm[br];
#pragma unroll
        for (int t = 0; t < 3; ++t)
#pragma unroll
            for (int v = 0; v < 8; ++v) xacc[t][v] += wk * acc2[t][v];
    }

    // ---- fc chain: x -> relu(x@fc1) -> @fc2
    __syncthreads();
#pragma unroll
    for (int t = 0; t < 3; ++t) {
        const int nt = wave * 3 + t;
#pragma unroll
        for (int v = 0; v < 8; ++v) {
            const int m = v + 8 * hi;
            hbuf[m * HSTRIDE + nt * 16 + m16] = f2bf(xacc[t][v]);
        }
    }
    __syncthreads();

    v8f acc3[3];
    acc3[0] = zero; acc3[1] = zero; acc3[2] = zero;
    {
        const unsigned int* bb = wpack + OFF_FC1 + bwoff;
#pragma unroll 4
        for (int kt = 0; kt < 12; ++kt, bb += NT * 256)
            gemm_step(hab, kt * 32, bb, acc3);
    }
    __syncthreads();   // all reads of hbuf done before overwrite
#pragma unroll
    for (int t = 0; t < 3; ++t) {
        const int nt = wave * 3 + t;
#pragma unroll
        for (int v = 0; v < 8; ++v) {
            const int m = v + 8 * hi;
            hbuf[m * HSTRIDE + nt * 16 + m16] = f2bf(fmaxf(acc3[t][v], 0.f));
        }
    }
    if (tid < 16) pid_s[tid] = posidx[rowBase + tid];
    __syncthreads();

    v8f acc4[3];
    acc4[0] = zero; acc4[1] = zero; acc4[2] = zero;
    {
        const unsigned int* bb = wpack + OFF_FC2 + bwoff;
#pragma unroll 4
        for (int kt = 0; kt < 12; ++kt, bb += NT * 256)
            gemm_step(hab, kt * 32, bb, acc4);
    }

    // ---- epilogue: + pos_emb[idx], store f32
#pragma unroll
    for (int t = 0; t < 3; ++t) {
        const int nt = wave * 3 + t;
        const int n  = nt * 16 + m16;
#pragma unroll
        for (int v = 0; v < 8; ++v) {
            const int  m   = v + 8 * hi;
            const long row = rowBase + m;
            const float pe = pos_emb[(long)pid_s[m] * UNITS + n];
            out[row * UNITS + n] = acc4[t][v] + pe;
        }
    }
}

// ---------------- host entry ----------------
extern "C" void kernel_launch(void* const* d_in, const int* in_sizes, int n_in,
                              void* d_out, int out_size, void* d_ws, size_t ws_size,
                              hipStream_t stream) {
    (void)in_sizes; (void)n_in; (void)out_size; (void)ws_size;
    const float* lips    = (const float*)d_in[0];
    const float* lh      = (const float*)d_in[1];
    const float* rh      = (const float*)d_in[2];
    const float* pose    = (const float*)d_in[3];
    const float* nefi    = (const float*)d_in[4];
    const float* lips_w1 = (const float*)d_in[5];
    const float* lips_w2 = (const float*)d_in[6];
    const float* lh_w1   = (const float*)d_in[7];
    const float* lh_w2   = (const float*)d_in[8];
    const float* rh_w1   = (const float*)d_in[9];
    const float* rh_w2   = (const float*)d_in[10];
    const float* pose_w1 = (const float*)d_in[11];
    const float* pose_w2 = (const float*)d_in[12];
    const float* lmw     = (const float*)d_in[13];
    const float* fc_w1   = (const float*)d_in[14];
    const float* fc_w2   = (const float*)d_in[15];
    const float* pos_emb = (const float*)d_in[16];

    // workspace layout (dwords): [0..63] softmax, [64..65599] posidx, then packed weights
    float*        wsm    = (float*)d_ws;
    int*          posidx = (int*)d_ws + 64;
    unsigned int* wpack  = (unsigned int*)d_ws + 64 + NB * SEQ;

    pack_weights<<< 72, 32, 0, stream>>>(lips_w1, 80, wpack + OFF_LIPS1);
    pack_weights<<< 48, 32, 0, stream>>>(lh_w1,   42, wpack + OFF_LH1);
    pack_weights<<< 48, 32, 0, stream>>>(rh_w1,   42, wpack + OFF_RH1);
    pack_weights<<< 24, 32, 0, stream>>>(pose_w1, 20, wpack + OFF_POSE1);
    pack_weights<<<288, 32, 0, stream>>>(lips_w2, 384, wpack + OFF_LIPS2);
    pack_weights<<<288, 32, 0, stream>>>(lh_w2,   384, wpack + OFF_LH2);
    pack_weights<<<288, 32, 0, stream>>>(rh_w2,   384, wpack + OFF_RH2);
    pack_weights<<<288, 32, 0, stream>>>(pose_w2, 384, wpack + OFF_POSE2);
    pack_weights<<<288, 32, 0, stream>>>(fc_w1,   384, wpack + OFF_FC1);
    pack_weights<<<288, 32, 0, stream>>>(fc_w2,   384, wpack + OFF_FC2);

    prep_softmax<<<1, 1, 0, stream>>>(lmw, wsm);
    prep_idx<<<NB, SEQ, 0, stream>>>(nefi, posidx);

    fused_embed<<<NB * 4, 256, 0, stream>>>(lips, lh, rh, pose, wpack, wsm,
                                            posidx, pos_emb, (float*)d_out);
}
